// MultiBoxLoss_66279935312450
// MI455X (gfx1250) — compile-verified, matrix operand order checked by reading, not run on hardware
//
#include <hip/hip_runtime.h>

typedef _Float16 v16h __attribute__((ext_vector_type(16)));
typedef float    v8f  __attribute__((ext_vector_type(8)));

#define THRESH 0.5f
#define VAR0 0.1f
#define VAR1 0.2f

// ---------------------------------------------------------------------------
// Kernel A: per-GT best prior (argmax IoU over P), first-index tie break.
// Also zero-initializes num_pos[b].
// ---------------------------------------------------------------------------
__global__ void best_prior_kernel(const float* __restrict__ priors,
                                  const float* __restrict__ tboxes,
                                  int* __restrict__ bpi,
                                  int* __restrict__ numpos,
                                  int P, int NOBJ) {
    const int b = blockIdx.x;
    const int t = threadIdx.x;
    if (t == 0) numpos[b] = 0;
    const int wave = t >> 5, lane = t & 31;
    const int nwaves = blockDim.x >> 5;
    const float4* pr4 = (const float4*)priors;
    const float* tb = tboxes + (size_t)b * NOBJ * 4;
    for (int g = wave; g < NOBJ; g += nwaves) {
        const float t0 = tb[g * 4 + 0], t1 = tb[g * 4 + 1];
        const float t2 = tb[g * 4 + 2], t3 = tb[g * 4 + 3];
        const float areaT = (t2 - t0) * (t3 - t1);
        float bestv = -1.0f;
        int   besti = 0;
        for (int p = lane; p < P; p += 32) {
            const float4 pr = pr4[p];
            const float p0 = pr.x - 0.5f * pr.z, p1 = pr.y - 0.5f * pr.w;
            const float p2 = pr.x + 0.5f * pr.z, p3 = pr.y + 0.5f * pr.w;
            const float iw = fmaxf(fminf(t2, p2) - fmaxf(t0, p0), 0.0f);
            const float ih = fmaxf(fminf(t3, p3) - fmaxf(t1, p1), 0.0f);
            const float inter = iw * ih;
            const float iou = inter / (areaT + pr.z * pr.w - inter);
            if (iou > bestv) { bestv = iou; besti = p; }
        }
        // wave32 argmax reduce; tie -> smaller prior index (matches jnp.argmax)
        for (int off = 16; off > 0; off >>= 1) {
            const float ov = __shfl_down(bestv, off);
            const int   oi = __shfl_down(besti, off);
            if (ov > bestv || (ov == bestv && oi < besti)) { bestv = ov; besti = oi; }
        }
        if (lane == 0) bpi[b * NOBJ + g] = besti;
    }
}

// ---------------------------------------------------------------------------
// Kernel B: fused matching + encode + SmoothL1 + CE.
// CE sum-of-exp over classes done with V_WMMA_F32_16X16X32_F16 (B = ones).
// Block = 256 threads handling 256 consecutive priors of batch b.
// Assumes 17 <= C <= 21 (reference: C = 21).
// ---------------------------------------------------------------------------
__global__ void match_ce_kernel(const float* __restrict__ loc_data,
                                const float* __restrict__ conf_data,
                                const float* __restrict__ priors,
                                const float* __restrict__ tboxes,
                                const int*   __restrict__ tlabels,
                                const int*   __restrict__ bpi,
                                int*   __restrict__ numpos,
                                float* __restrict__ lossmine,
                                float* __restrict__ locpart,
                                float* __restrict__ ceppart,
                                int P, int C, int NOBJ) {
    const int b    = blockIdx.y;
    const int base = blockIdx.x * 256;
    const int t    = threadIdx.x;

    __shared__ float s_truth[64 * 4];
    __shared__ int   s_lab[64];
    __shared__ int   s_bpi[64];
    __shared__ int   s_conf[256];
    __shared__ float s_red[256];
    __shared__ int   s_ired[8];

    if (t < NOBJ) {
        s_lab[t] = tlabels[b * NOBJ + t];
        s_bpi[t] = bpi[b * NOBJ + t];
    }
    if (t < NOBJ * 4) s_truth[t] = tboxes[(size_t)b * NOBJ * 4 + t];
    __syncthreads();

    // ---------------- part 1: matching + encode + smooth L1 ----------------
    const int  p     = base + t;
    const bool valid = (p < P);
    const int  pp    = valid ? p : 0;
    const float4 pr  = ((const float4*)priors)[pp];
    const float cx = pr.x, cy = pr.y, w = pr.z, h = pr.w;
    const float p0 = cx - 0.5f * w, p1 = cy - 0.5f * h;
    const float p2 = cx + 0.5f * w, p3 = cy + 0.5f * h;
    const float areaP = w * h;

    float bestv = -1.0f;
    int   bidx  = 0;
    for (int n = 0; n < NOBJ; ++n) {
        const float t0 = s_truth[n * 4 + 0], t1 = s_truth[n * 4 + 1];
        const float t2 = s_truth[n * 4 + 2], t3 = s_truth[n * 4 + 3];
        const float iw = fmaxf(fminf(t2, p2) - fmaxf(t0, p0), 0.0f);
        const float ih = fmaxf(fminf(t3, p3) - fmaxf(t1, p1), 0.0f);
        const float inter = iw * ih;
        const float iou = inter / ((t2 - t0) * (t3 - t1) + areaP - inter);
        if (iou > bestv) { bestv = iou; bidx = n; }  // strict > keeps first max
    }
    for (int n = 0; n < NOBJ; ++n)          // scatter override, last write wins
        if (s_bpi[n] == p) { bestv = 2.0f; bidx = n; }

    int conf = (bestv < THRESH) ? 0 : (s_lab[bidx] + 1);
    if (!valid) conf = 0;
    s_conf[t] = conf;

    float locAcc = 0.0f;
    int   posc   = 0;
    if (conf > 0) {
        const float m0 = s_truth[bidx * 4 + 0], m1 = s_truth[bidx * 4 + 1];
        const float m2 = s_truth[bidx * 4 + 2], m3 = s_truth[bidx * 4 + 3];
        const float gx = ((m0 + m2) * 0.5f - cx) / (VAR0 * w);
        const float gy = ((m1 + m3) * 0.5f - cy) / (VAR0 * h);
        const float gw = __logf((m2 - m0) / w) / VAR1;
        const float gh = __logf((m3 - m1) / h) / VAR1;
        const float4 l4 = ((const float4*)loc_data)[(size_t)b * P + p];
        const float d0 = l4.x - gx, d1 = l4.y - gy, d2 = l4.z - gw, d3 = l4.w - gh;
        const float a0 = fabsf(d0), a1 = fabsf(d1), a2 = fabsf(d2), a3 = fabsf(d3);
        locAcc = (a0 < 1.0f ? 0.5f * d0 * d0 : a0 - 0.5f)
               + (a1 < 1.0f ? 0.5f * d1 * d1 : a1 - 0.5f)
               + (a2 < 1.0f ? 0.5f * d2 * d2 : a2 - 0.5f)
               + (a3 < 1.0f ? 0.5f * d3 * d3 : a3 - 0.5f);
        posc = 1;
    }

    // integer pos count: wave reduce -> shared -> single atomicAdd (deterministic)
    {
        int c = posc;
        for (int off = 16; off > 0; off >>= 1) c += __shfl_down(c, off);
        if ((t & 31) == 0) s_ired[t >> 5] = c;
    }
    __syncthreads();
    if (t == 0) {
        int tot = 0;
        for (int i = 0; i < 8; ++i) tot += s_ired[i];
        atomicAdd(&numpos[b], tot);
    }

    // ---------------- part 2: CE with WMMA sum-of-exp over classes ----------
    float ceAcc = 0.0f;
    {
        v16h bones;
#pragma unroll
        for (int e = 0; e < 16; ++e) bones[e] = (_Float16)1.0f;

        const int wv       = t >> 5;
        const int lane     = t & 31;
        const int laneHalf = lane >> 4;       // 0 or 1
        const int lm16     = lane & 15;
        const int ofs      = laneHalf * 8;    // contiguous base offset per half

        for (int r = 0; r < 2; ++r) {
            const int m16 = base + 32 * wv + 16 * r;   // wave-uniform
            if (m16 >= P) continue;
            const int  q  = m16 + lm16;
            const bool vq = (q < P);
            const float* row = conf_data + ((size_t)b * P + (vq ? q : 0)) * C;

            // Unconditional contiguous loads -> vectorizable global_load_b128.
            // chunk0: row[ofs .. ofs+7]  (K = e or e+8, always < 16 <= C)
            // chunk1: row[16 .. 20]      (K = 16..20; only used by laneHalf==0)
            float v0[8];
#pragma unroll
            for (int i = 0; i < 8; ++i) v0[i] = row[ofs + i];
            float v1[5];
#pragma unroll
            for (int i = 0; i < 5; ++i) v1[i] = row[16 + i];

            // 16-bit A layout, 16x32 (MxK):
            //   lanes 0-15 : e -> K = e (e<8) else e+8   (K in {0..7,16..23})
            //   lanes 16-31: e -> K = e+8 (e<8) else e+16 (K in {8..15,24..31})
            v16h a;
#pragma unroll
            for (int e = 0; e < 8; ++e)
                a[e] = vq ? (_Float16)__expf(v0[e]) : (_Float16)0.0f;
#pragma unroll
            for (int e = 8; e < 16; ++e) {
                const int K = 8 + e + 8 * laneHalf;    // laneHalf0: 16..23, laneHalf1: 24..31
                const bool use = vq && (K < C);
                const float val = (e < 13) ? v1[e - 8] : 0.0f;
                a[e] = use ? (_Float16)__expf(val) : (_Float16)0.0f;
            }

            v8f c;
#pragma unroll
            for (int e = 0; e < 8; ++e) c[e] = 0.0f;
            // D[m][n] = sum_k A[m][k]  (B = ones) -> per-prior sum of exp
            c = __builtin_amdgcn_wmma_f32_16x16x32_f16(
                    false, a, false, bones, (short)0, c, false, false);

            // D layout: lanes 0-15 hold M=0..7 in c[0..7]; lanes 16-31 hold M=8..15.
            // Branchless 3-level select tree (all lanes, full EXEC -> v_cndmask).
            const int j = lane & 7;
            const float s01 = ((j & 1) != 0) ? c[1] : c[0];
            const float s23 = ((j & 1) != 0) ? c[3] : c[2];
            const float s45 = ((j & 1) != 0) ? c[5] : c[4];
            const float s67 = ((j & 1) != 0) ? c[7] : c[6];
            const float sA  = ((j & 2) != 0) ? s23 : s01;
            const float sB  = ((j & 2) != 0) ? s67 : s45;
            const float s   = ((j & 4) != 0) ? sB : sA;

            const bool resp = (lane & 15) < 8;
            const int  m    = (lane & 7) + 8 * laneHalf;
            const int  qr   = m16 + m;
            if (resp && qr < P) {
                const int   ct = s_conf[qr - base];
                const float xt = conf_data[((size_t)b * P + qr) * C + ct];
                const float ce = __logf(s) - xt;
                const bool ispos = (ct > 0);
                ceAcc += ispos ? ce : 0.0f;
                lossmine[(size_t)b * P + qr] = ispos ? 0.0f : fmaxf(ce, 0.0f);
            }
        }
    }

    // ---------------- deterministic block reductions ----------------
    s_red[t] = locAcc;
    __syncthreads();
    for (int s = 128; s > 0; s >>= 1) { if (t < s) s_red[t] += s_red[t + s]; __syncthreads(); }
    if (t == 0) locpart[blockIdx.y * gridDim.x + blockIdx.x] = s_red[0];
    __syncthreads();
    s_red[t] = ceAcc;
    __syncthreads();
    for (int s = 128; s > 0; s >>= 1) { if (t < s) s_red[t] += s_red[t + s]; __syncthreads(); }
    if (t == 0) ceppart[blockIdx.y * gridDim.x + blockIdx.x] = s_red[0];
}

// ---------------------------------------------------------------------------
// Kernel C: per-batch exact top-K sum of loss_mine (hard negative mining)
// via 4-level radix select on non-negative float bit patterns.
// ---------------------------------------------------------------------------
__global__ void topk_kernel(const float* __restrict__ lossmine,
                            const int*   __restrict__ numpos,
                            float* __restrict__ topk,
                            int P, int NPR) {
    const int b = blockIdx.x;
    const int t = threadIdx.x;
    const float* lm = lossmine + (size_t)b * P;

    __shared__ unsigned cnt[256];
    __shared__ unsigned s_pref;
    __shared__ int      s_krem;
    __shared__ float    s_red[256];

    int K = NPR * numpos[b];
    if (K > P - 1) K = P - 1;
    if (K <= 0) { if (t == 0) topk[b] = 0.0f; return; }  // uniform exit

    if (t == 0) { s_pref = 0u; s_krem = K; }
    __syncthreads();

    for (int l = 0; l < 4; ++l) {
        cnt[t] = 0u;
        __syncthreads();
        const unsigned pref = s_pref;
        const int shift = 24 - 8 * l;
        for (int i = t; i < P; i += 256) {
            const unsigned u = __float_as_uint(lm[i]);
            const bool sel = (l == 0) || (((u ^ pref) >> (shift + 8)) == 0u);
            if (sel) atomicAdd(&cnt[(u >> shift) & 255u], 1u);
        }
        __syncthreads();
        if (t == 0) {
            int cum = 0;
            const int krem = s_krem;
            for (int bin = 255; bin >= 0; --bin) {
                const int nc = cum + (int)cnt[bin];
                if (nc >= krem) {
                    s_pref = pref | ((unsigned)bin << shift);
                    s_krem = krem - cum;
                    break;
                }
                cum = nc;
            }
        }
        __syncthreads();
    }

    const float vf   = __uint_as_float(s_pref);
    const int   krem = s_krem;
    float acc = 0.0f;
    for (int i = t; i < P; i += 256) {
        const float f = lm[i];
        if (f > vf) acc += f;
    }
    s_red[t] = acc;
    __syncthreads();
    for (int s = 128; s > 0; s >>= 1) { if (t < s) s_red[t] += s_red[t + s]; __syncthreads(); }
    if (t == 0) topk[b] = s_red[0] + (float)krem * vf;
}

// ---------------------------------------------------------------------------
// Kernel D: finalize -> out[0] = loss_l / N, out[1] = loss_c / N
// ---------------------------------------------------------------------------
__global__ void finalize_kernel(const float* __restrict__ locpart,
                                const float* __restrict__ ceppart,
                                int nparts,
                                const float* __restrict__ topk,
                                const int*   __restrict__ numpos,
                                int B,
                                float* __restrict__ out) {
    const int t = threadIdx.x;
    __shared__ float s_red[256];

    float a = 0.0f;
    for (int i = t; i < nparts; i += 256) a += locpart[i];
    s_red[t] = a;
    __syncthreads();
    for (int s = 128; s > 0; s >>= 1) { if (t < s) s_red[t] += s_red[t + s]; __syncthreads(); }
    const float loss_l = s_red[0];
    __syncthreads();

    a = 0.0f;
    for (int i = t; i < nparts; i += 256) a += ceppart[i];
    for (int i = t; i < B; i += 256) a += topk[i];
    s_red[t] = a;
    __syncthreads();
    for (int s = 128; s > 0; s >>= 1) { if (t < s) s_red[t] += s_red[t + s]; __syncthreads(); }
    const float loss_c = s_red[0];
    __syncthreads();

    int np = 0;
    for (int i = t; i < B; i += 256) np += numpos[i];
    s_red[t] = (float)np;
    __syncthreads();
    for (int s = 128; s > 0; s >>= 1) { if (t < s) s_red[t] += s_red[t + s]; __syncthreads(); }
    if (t == 0) {
        const float N = fmaxf(s_red[0], 1.0f);
        out[0] = loss_l / N;
        out[1] = loss_c / N;
    }
}

// ---------------------------------------------------------------------------
extern "C" void kernel_launch(void* const* d_in, const int* in_sizes, int n_in,
                              void* d_out, int out_size, void* d_ws, size_t ws_size,
                              hipStream_t stream) {
    (void)n_in; (void)out_size; (void)ws_size;
    const float* loc_data  = (const float*)d_in[0];
    const float* conf_data = (const float*)d_in[1];
    const float* priors    = (const float*)d_in[2];
    const float* tboxes    = (const float*)d_in[3];
    const int*   tlabels   = (const int*)d_in[4];
    float* out = (float*)d_out;

    const int P    = in_sizes[2] / 4;
    const int B    = in_sizes[0] / (P * 4);
    const int C    = in_sizes[1] / (B * P);
    const int NOBJ = in_sizes[4] / B;

    const int gridX  = (P + 255) / 256;
    const int nparts = gridX * B;

    char* wsp = (char*)d_ws;
    auto wsalloc = [&](size_t bytes) -> char* {
        char* r = wsp;
        wsp += (bytes + 255) & ~(size_t)255;
        return r;
    };
    float* lossmine = (float*)wsalloc((size_t)B * P * sizeof(float));
    int*   bpi      = (int*)  wsalloc((size_t)B * NOBJ * sizeof(int));
    int*   numpos   = (int*)  wsalloc((size_t)B * sizeof(int));
    float* locpart  = (float*)wsalloc((size_t)nparts * sizeof(float));
    float* ceppart  = (float*)wsalloc((size_t)nparts * sizeof(float));
    float* topk     = (float*)wsalloc((size_t)B * sizeof(float));

    best_prior_kernel<<<B, 256, 0, stream>>>(priors, tboxes, bpi, numpos, P, NOBJ);
    match_ce_kernel<<<dim3(gridX, B), 256, 0, stream>>>(
        loc_data, conf_data, priors, tboxes, tlabels,
        bpi, numpos, lossmine, locpart, ceppart, P, C, NOBJ);
    topk_kernel<<<B, 256, 0, stream>>>(lossmine, numpos, topk, P, 3);
    finalize_kernel<<<1, 256, 0, stream>>>(locpart, ceppart, nparts, topk, numpos, B, out);
}
